// CUBASpikingCNN_50053548868257
// MI455X (gfx1250) — matrix operand order repaired and checked by reading, not compiled, not run on hardware
//
#include <hip/hip_runtime.h>

typedef __attribute__((ext_vector_type(16))) _Float16 v16h;
typedef __attribute__((ext_vector_type(8)))  _Float16 v8h;
typedef __attribute__((ext_vector_type(8)))  float    v8f;
typedef __attribute__((__vector_size__(4 * sizeof(int)))) int v4i;

#if defined(__has_builtin)
#if __has_builtin(__builtin_amdgcn_global_load_async_to_lds_b128)
#define USE_ASYNC_LDS 1
#endif
#endif

namespace snn {
constexpr int B = 32, T = 200, H = 10, W = 11, NCLASS = 4;
constexpr float VTH = 0.3f, TH_AMP = 0.05f, TH_DECAY = 0.9f, BASE_TH = 0.3f;

constexpr int C1c = 64, H1 = 8, W1 = 9;    // conv1 out
constexpr int C2c = 128, H2 = 6, W2 = 7;   // conv2 out
constexpr int N1 = B * C1c * H1 * W1;      // 147456
constexpr int N2 = B * C2c * H2 * W2;      // 172032
constexpr int ND = B * 256;                // 8192
constexpr int M2 = B * H2 * W2;            // 1344 (GEMM2 M)
constexpr int K2 = C1c * 9;                // 576  (GEMM2 K)
constexpr int K3 = C2c * 9;                // 1152 (GEMM3 K)
constexpr int NP = B * K3;                 // pooled, flattened = im2col for conv3

constexpr size_t AL(size_t x) { return (x + 255) & ~(size_t)255; }
// f32 state / staging
constexpr size_t O_C1  = 0;
constexpr size_t O_V1  = AL(O_C1  + (size_t)N1 * 4);
constexpr size_t O_C2  = AL(O_V1  + (size_t)N1 * 4);
constexpr size_t O_V2  = AL(O_C2  + (size_t)N2 * 4);
constexpr size_t O_C3  = AL(O_V2  + (size_t)N2 * 4);
constexpr size_t O_V3  = AL(O_C3  + (size_t)ND * 4);
constexpr size_t O_CTC = AL(O_V3  + (size_t)ND * 4);
constexpr size_t O_VTC = AL(O_CTC + (size_t)ND * 4);
constexpr size_t O_CR  = AL(O_VTC + (size_t)ND * 4);
constexpr size_t O_VR  = AL(O_CR  + (size_t)ND * 4);
constexpr size_t O_TH  = AL(O_VR  + (size_t)ND * 4);
constexpr size_t O_CF  = AL(O_TH  + (size_t)ND * 4);
constexpr size_t O_VF  = AL(O_CF  + (size_t)ND * 4);
constexpr size_t O_Y2  = AL(O_VF  + (size_t)ND * 4);
constexpr size_t O_Y3  = AL(O_Y2  + (size_t)M2 * 128 * 4);
constexpr size_t O_YTC = AL(O_Y3  + (size_t)ND * 4);
constexpr size_t O_YR  = AL(O_YTC + (size_t)ND * 4);
constexpr size_t O_YF  = AL(O_YR  + (size_t)ND * 4);
constexpr size_t O_ACC = AL(O_YF  + (size_t)ND * 4);
// f16 activations / weights (K-major Bt layout)
constexpr size_t O_S1   = AL(O_ACC  + 128 * 4);
constexpr size_t O_A2   = AL(O_S1   + (size_t)N1 * 2);
constexpr size_t O_S2   = AL(O_A2   + (size_t)M2 * K2 * 2);
constexpr size_t O_P    = AL(O_S2   + (size_t)N2 * 2);
constexpr size_t O_BUF  = AL(O_P    + (size_t)NP * 2);
constexpr size_t O_STC  = AL(O_BUF  + (size_t)3 * ND * 2);
constexpr size_t O_SR   = AL(O_STC  + (size_t)ND * 2);
constexpr size_t O_SF   = AL(O_SR   + (size_t)ND * 2);
constexpr size_t O_BT2  = AL(O_SF   + (size_t)ND * 2);
constexpr size_t O_BT3  = AL(O_BT2  + (size_t)K2 * 128 * 2);
constexpr size_t O_BTTC = AL(O_BT3  + (size_t)K3 * 256 * 2);
constexpr size_t O_BTR1 = AL(O_BTTC + (size_t)768 * 256 * 2);
constexpr size_t O_BTRE = AL(O_BTR1 + (size_t)65536 * 2);
constexpr size_t O_BTF1 = AL(O_BTRE + (size_t)65536 * 2);
constexpr size_t O_SYNC = AL(O_BTF1 + (size_t)65536 * 2);
} // namespace snn

// ---------------- grid-wide barrier (persistent kernel) ----------------
__device__ __forceinline__ void gsync(unsigned* bar, int nwg) {
  __syncthreads();
  if (threadIdx.x == 0) {
    volatile unsigned* gen = bar + 1;
    __threadfence();
    unsigned g = *gen;
    if (atomicAdd(bar, 1u) + 1u == (unsigned)nwg) {
      atomicExch(bar, 0u);
      __threadfence();
      atomicAdd((unsigned*)(bar + 1), 1u);
    } else {
      while (*gen == g) __builtin_amdgcn_s_sleep(2);
    }
    __threadfence();
  }
  __syncthreads();
}

// -------- WMMA f16 fragment loaders (ISA 7.12.2 layouts, wave32) --------
// A (16x32 f16): lane m=lane&15, kh=lane>>4; halves 0-7 = K(8*kh..),
// halves 8-15 = K(16+8*kh..)
__device__ __forceinline__ v16h load_a_frag(const _Float16* A, int lda,
                                            int m0, int k0) {
  int lane = threadIdx.x & 31;
  int m = lane & 15, kh = lane >> 4;
  const _Float16* p = A + (size_t)(m0 + m) * lda + k0 + 8 * kh;
  union { v16h v; v8h h[2]; } u;
  u.h[0] = *(const v8h*)p;
  u.h[1] = *(const v8h*)(p + 16);
  return u.v;
}
// B (32x16 f16, Bt stored K-major [K][N]): lane n=lane&15, kh=lane>>4;
// halves 0-7 = K(8*kh..), halves 8-15 = K(16+8*kh..)
__device__ __forceinline__ v16h load_b_frag(const _Float16* Bt, int ldb,
                                            int k0, int n0) {
  int lane = threadIdx.x & 31;
  const _Float16* p = Bt + (size_t)(k0 + 8 * (lane >> 4)) * ldb + n0 + (lane & 15);
  union { v16h v; v8h h[2]; } u;
  for (int s = 0; s < 8; ++s) {
    u.v[s] = p[(size_t)s * ldb];
    u.v[8 + s] = p[(size_t)(16 + s) * ldb];
  }
  return u.v;
}
// C/D (16x16 f32): lane n=lane&15, rows m = r + 8*(lane>>4)
__device__ __forceinline__ void store_tile(float* C, int ldc, int m0, int n0,
                                           v8f acc) {
  int lane = threadIdx.x & 31;
  int n = lane & 15, mh = lane >> 4;
#pragma unroll
  for (int r = 0; r < 8; ++r)
    C[(size_t)(m0 + r + 8 * mh) * ldc + n0 + n] = acc[r];
}

__device__ __forceinline__ v8f wmma_kloop(const _Float16* A, int lda,
                                          const _Float16* Bt, int ldb,
                                          int m0, int n0, int K, v8f acc) {
  int lane = threadIdx.x & 31;
  for (int k0 = 0; k0 < K; k0 += 32) {
    // speculative prefetch of the next B-tile rows (global_prefetch_b8)
    __builtin_prefetch(Bt + (size_t)(k0 + 32 + lane) * ldb + n0, 0, 1);
    v16h a = load_a_frag(A, lda, m0, k0);
    v16h b = load_b_frag(Bt, ldb, k0, n0);
    acc = __builtin_amdgcn_wmma_f32_16x16x32_f16(false, a, false, b,
                                                 (short)0, acc, false, false);
  }
  return acc;
}

__device__ __forceinline__ void gemm(const _Float16* A, int lda,
                                     const _Float16* Bt, int N, float* C,
                                     int M, int Nn, int K, int wid, int nw) {
  int mt = M >> 4, nt = Nn >> 4;
  for (int tt = wid; tt < mt * nt; tt += nw) {
    int m0 = (tt / nt) << 4, n0 = (tt % nt) << 4;
    v8f acc = {};
    acc = wmma_kloop(A, lda, Bt, N, m0, n0, K, acc);
    store_tile(C, Nn, m0, n0, acc);
  }
}

struct SnnParams {
  const float* in[33];
  float* out;
  char* ws;
  int nwg;
};

__global__ __launch_bounds__(256) void snn_persistent(SnnParams p) {
  using namespace snn;
  const int tid = threadIdx.x;
  const int gtid = blockIdx.x * blockDim.x + tid;
  const int nth = p.nwg * 256;
  const int wid = gtid >> 5, nw = nth >> 5;

  // LDS-resident quarter of BT2 (conv2 B panel): 576 x 32 f16 = 36 KB.
  // Constant over all 200 timesteps -> staged once, read via ds_load.
  __shared__ _Float16 lB[K2 * 32];

  char* ws = p.ws;
  float* c1S = (float*)(ws + O_C1);  float* v1S = (float*)(ws + O_V1);
  float* c2S = (float*)(ws + O_C2);  float* v2S = (float*)(ws + O_V2);
  float* c3S = (float*)(ws + O_C3);  float* v3S = (float*)(ws + O_V3);
  float* ctcS = (float*)(ws + O_CTC); float* vtcS = (float*)(ws + O_VTC);
  float* crS = (float*)(ws + O_CR);  float* vrS = (float*)(ws + O_VR);
  float* thS = (float*)(ws + O_TH);
  float* cfS = (float*)(ws + O_CF);  float* vfS = (float*)(ws + O_VF);
  float* Y2 = (float*)(ws + O_Y2);   float* Y3 = (float*)(ws + O_Y3);
  float* YTC = (float*)(ws + O_YTC); float* YR = (float*)(ws + O_YR);
  float* YF = (float*)(ws + O_YF);   float* ACC = (float*)(ws + O_ACC);
  _Float16* S1 = (_Float16*)(ws + O_S1);   _Float16* A2 = (_Float16*)(ws + O_A2);
  _Float16* S2 = (_Float16*)(ws + O_S2);   _Float16* P  = (_Float16*)(ws + O_P);
  _Float16* BUF = (_Float16*)(ws + O_BUF); _Float16* STC = (_Float16*)(ws + O_STC);
  _Float16* SR = (_Float16*)(ws + O_SR);   _Float16* SF = (_Float16*)(ws + O_SF);
  _Float16* BT2 = (_Float16*)(ws + O_BT2); _Float16* BT3 = (_Float16*)(ws + O_BT3);
  _Float16* BTTC = (_Float16*)(ws + O_BTTC);
  _Float16* BTR1 = (_Float16*)(ws + O_BTR1);
  _Float16* BTRE = (_Float16*)(ws + O_BTRE);
  _Float16* BTF1 = (_Float16*)(ws + O_BTF1);
  unsigned* bar = (unsigned*)(ws + O_SYNC);

  const float* xin = p.in[0];
  const float* w1 = p.in[1];  const float* b1 = p.in[2];
  const float* w2 = p.in[3];  const float* b2 = p.in[4];
  const float* w3 = p.in[5];  const float* b3 = p.in[6];
  const float* tcw = p.in[7]; const float* tcb = p.in[8];
  const float* r1w = p.in[9]; const float* r1b = p.in[10];
  const float* recw = p.in[11]; const float* recb = p.in[12];
  const float* f1w = p.in[13]; const float* f1b = p.in[14];
  const float* fcw = p.in[15]; const float* fcb = p.in[16];
  const float* c1cd = p.in[17]; const float* c1vd = p.in[18];
  const float* c2cd = p.in[19]; const float* c2vd = p.in[20];
  const float* c3cd = p.in[21]; const float* c3vd = p.in[22];
  const float* tccd = p.in[23]; const float* tcvd = p.in[24];
  const float* r1cd = p.in[25]; const float* r1vd = p.in[26];
  const float* f1cd = p.in[27]; const float* f1vd = p.in[28];
  const float* tsw = p.in[29];
  const float* tcm = p.in[30]; const float* r1m = p.in[31];
  const float* f1m = p.in[32];

  // ---------------- init: zero state, prep f16 K-major weights ----------
  for (int i = gtid; i < N1; i += nth) {
    c1S[i] = 0.f; v1S[i] = 0.f; S1[i] = (_Float16)0.f;
  }
  for (int i = gtid; i < N2; i += nth) {
    c2S[i] = 0.f; v2S[i] = 0.f; S2[i] = (_Float16)0.f;
  }
  for (int i = gtid; i < ND; i += nth) {
    c3S[i] = 0.f; v3S[i] = 0.f; ctcS[i] = 0.f; vtcS[i] = 0.f;
    crS[i] = 0.f; vrS[i] = 0.f; cfS[i] = 0.f; vfS[i] = 0.f;
    thS[i] = VTH;
    STC[i] = (_Float16)0.f; SR[i] = (_Float16)0.f; SF[i] = (_Float16)0.f;
    BUF[i] = (_Float16)0.f; BUF[ND + i] = (_Float16)0.f;
    BUF[2 * ND + i] = (_Float16)0.f;
  }
  for (int i = gtid; i < B * NCLASS; i += nth) ACC[i] = 0.f;
  for (int i = gtid; i < K2 * 128; i += nth) {
    int k = i >> 7, n = i & 127;
    BT2[i] = (_Float16)w2[n * K2 + k];
  }
  for (int i = gtid; i < K3 * 256; i += nth) {
    int k = i >> 8, n = i & 255;
    BT3[i] = (_Float16)w3[n * K3 + k];
  }
  for (int i = gtid; i < 768 * 256; i += nth) {
    int kk = i >> 8, o = i & 255;
    int k = kk >> 8, ii = kk & 255;
    BTTC[i] = (_Float16)tcw[k * 65536 + o * 256 + ii];
  }
  for (int i = gtid; i < 65536; i += nth) {
    int ii = i >> 8, o = i & 255;
    BTR1[i] = (_Float16)r1w[o * 256 + ii];
    BTRE[i] = (_Float16)recw[o * 256 + ii];
    BTF1[i] = (_Float16)f1w[o * 256 + ii];
  }
  gsync(bar, p.nwg);  // BT2 ready in global before LDS staging

  // ---- stage this WG's N-quarter of BT2 into LDS (async-to-LDS path) ----
  {
    const int q = blockIdx.x & 3;          // which 32-column quarter
    // 16-byte chunks: 576 rows * 4 chunks/row
    for (int c = tid; c < K2 * 4; c += 256) {
      int k = c >> 2, j8 = (c & 3) * 8;
      const _Float16* src = BT2 + (size_t)k * 128 + q * 32 + j8;
      _Float16* dst = lB + k * 32 + j8;
#ifdef USE_ASYNC_LDS
      __builtin_amdgcn_global_load_async_to_lds_b128((v4i*)src, (v4i*)dst,
                                                     0, 0);
#else
      *(v8h*)dst = *(const v8h*)src;
#endif
    }
#if defined(__has_builtin) && __has_builtin(__builtin_amdgcn_s_wait_asynccnt)
    __builtin_amdgcn_s_wait_asynccnt(0);
#else
    asm volatile("s_wait_asynccnt 0x0" ::: "memory");
#endif
    __syncthreads();
  }

  // ---------------- time loop ----------------
  for (int t = 0; t < T; ++t) {
    // A: conv1 (K=9, VALU) fused with LIF1
    for (int e = gtid; e < N1; e += nth) {
      int b = e / (C1c * H1 * W1);
      int r = e % (C1c * H1 * W1);
      int c = r / (H1 * W1);
      int yx = r % (H1 * W1);
      int y = yx / W1, x = yx % W1;
      const float* xb = xin + ((size_t)b * T + t) * (H * W);
      const float* wk = w1 + c * 9;
      float o = b1[c];
#pragma unroll
      for (int dy = 0; dy < 3; ++dy)
#pragma unroll
        for (int dx = 0; dx < 3; ++dx)
          o += xb[(y + dy) * W + (x + dx)] * wk[dy * 3 + dx];
      float cc = c1cd[r] * c1S[e] + o;
      float so = (float)S1[e];
      float vv = c1vd[r] * v1S[e] * (1.f - so) + cc;
      c1S[e] = cc; v1S[e] = vv;
      S1[e] = (_Float16)((vv > VTH) ? 1.f : 0.f);
    }
    gsync(bar, p.nwg);

    // B: im2col for conv2 (A2[m][k], m=b*42+y*7+x, k=c*9+dy*3+dx)
    for (int i = gtid; i < M2 * K2; i += nth) {
      int m = i / K2, k = i % K2;
      int b = m / 42, yx = m % 42;
      int y = yx / 7, x = yx % 7;
      int c = k / 9, d = k % 9;
      int dy = d / 3, dx = d % 3;
      A2[i] = S1[b * (C1c * H1 * W1) + c * (H1 * W1) + (y + dy) * W1 + (x + dx)];
    }
    gsync(bar, p.nwg);

    // C: conv2 WMMA GEMM 1344 x 128 x 576, B fragments from LDS.
    // 8 WGs share quarter q = bid&3; they cover 84 m-tiles x 2 n-tiles.
    {
      const int q = blockIdx.x & 3;
      const int wq = (blockIdx.x >> 2) * 8 + (tid >> 5);  // 0..63 in quarter
      const int nwq = (p.nwg >> 2) * 8;
      const int lane = tid & 31;
      for (int tt = wq; tt < 84 * 2; tt += nwq) {
        int m0 = (tt >> 1) << 4;
        int n0q = (tt & 1) << 4;
        v8f acc = {};
        for (int k0 = 0; k0 < K2; k0 += 32) {
          v16h a = load_a_frag(A2, K2, m0, k0);
          const _Float16* pb = lB + (k0 + 8 * (lane >> 4)) * 32 + n0q + (lane & 15);  // ds_load path
          union { v16h v; v8h h[2]; } ub;
          for (int s = 0; s < 8; ++s) {
            ub.v[s] = pb[s * 32];
            ub.v[8 + s] = pb[(16 + s) * 32];
          }
          acc = __builtin_amdgcn_wmma_f32_16x16x32_f16(
              false, a, false, ub.v, (short)0, acc, false, false);
        }
        store_tile(Y2, 128, m0, q * 32 + n0q, acc);
      }
    }
    gsync(bar, p.nwg);

    // D: LIF2 (scatter GEMM-layout -> NCHW state)
    for (int i = gtid; i < M2 * 128; i += nth) {
      int m = i >> 7, n = i & 127;
      int b = m / 42, yx = m % 42;
      int e = b * (C2c * H2 * W2) + n * (H2 * W2) + yx;
      int di = n * (H2 * W2) + yx;
      float cc = c2cd[di] * c2S[e] + (Y2[i] + b2[n]);
      float so = (float)S2[e];
      float vv = c2vd[di] * v2S[e] * (1.f - so) + cc;
      c2S[e] = cc; v2S[e] = vv;
      S2[e] = (_Float16)((vv > VTH) ? 1.f : 0.f);
    }
    gsync(bar, p.nwg);

    // E: 2x2 avg pool (exact in f16: spike sums * 0.25)
    for (int i = gtid; i < NP; i += nth) {
      int b = i / K3, r = i % K3;
      int c = r / 9, q2 = r % 9;
      int py = q2 / 3, px = q2 % 3;
      const _Float16* s =
          S2 + b * (C2c * H2 * W2) + c * (H2 * W2) + (2 * py) * W2 + 2 * px;
      float sum = (float)s[0] + (float)s[1] + (float)s[W2] + (float)s[W2 + 1];
      P[i] = (_Float16)(0.25f * sum);
    }
    gsync(bar, p.nwg);

    // F: conv3 as WMMA GEMM 32 x 256 x 1152 (pooled tensor IS the im2col)
    gemm(P, K3, BT3, 256, Y3, B, 256, K3, wid, nw);
    gsync(bar, p.nwg);

    // G: LIF3 -> rotating spike buffer slot t%3 (replaces concatenate-shift)
    {
      int snew = t % 3, sold = (t + 2) % 3;
      for (int i = gtid; i < ND; i += nth) {
        int o = i & 255;
        float cc = c3cd[o] * c3S[i] + (Y3[i] + b3[o]);
        float so = (float)BUF[sold * ND + i];
        float vv = c3vd[o] * v3S[i] * (1.f - so) + cc;
        c3S[i] = cc; v3S[i] = vv;
        BUF[snew * ND + i] = (_Float16)((vv > VTH) ? 1.f : 0.f);
      }
    }
    gsync(bar, p.nwg);

    // H: temporal conv = 3 accumulated WMMA GEMMs (chunk k uses slot (t+k+1)%3)
    for (int tt = wid; tt < 2 * 16; tt += nw) {
      int m0 = (tt / 16) << 4, n0 = (tt & 15) << 4;
      v8f acc = {};
#pragma unroll
      for (int k = 0; k < 3; ++k) {
        const _Float16* Ak = BUF + ((t + k + 1) % 3) * ND;
        acc = wmma_kloop(Ak, 256, BTTC + k * 65536, 256, m0, n0, 256, acc);
      }
      store_tile(YTC, 256, m0, n0, acc);
    }
    gsync(bar, p.nwg);

    // I: LIF tc (bias_on windowing + mask)
    {
      int k0b = (t >= 2) ? 0 : (2 - t);
      for (int i = gtid; i < ND; i += nth) {
        int o = i & 255;
        float bias = 0.f;
        for (int k = k0b; k < 3; ++k) bias += tcb[k * 256 + o];
        float cc = tccd[o] * ctcS[i] + YTC[i] + bias;
        cc *= tcm[i];
        float so = (float)STC[i];
        float vv = tcvd[o] * vtcS[i] * (1.f - so) + cc;
        ctcS[i] = cc; vtcS[i] = vv;
        STC[i] = (_Float16)((vv > VTH) ? 1.f : 0.f);
      }
    }
    gsync(bar, p.nwg);

    // J: recurrent layer = 2 accumulated WMMA GEMMs (stc@r1_w.T + sr@rec_w.T)
    for (int tt = wid; tt < 2 * 16; tt += nw) {
      int m0 = (tt / 16) << 4, n0 = (tt & 15) << 4;
      v8f acc = {};
      acc = wmma_kloop(STC, 256, BTR1, 256, m0, n0, 256, acc);
      acc = wmma_kloop(SR, 256, BTRE, 256, m0, n0, 256, acc);
      store_tile(YR, 256, m0, n0, acc);
    }
    gsync(bar, p.nwg);

    // K: LIF r with adaptive threshold
    for (int i = gtid; i < ND; i += nth) {
      int o = i & 255;
      float cc = r1cd[o] * crS[i] + YR[i] + r1b[o] + recb[o];
      cc *= r1m[i];
      float so = (float)SR[i];
      float vv = r1vd[o] * vrS[i] * (1.f - so) + cc;
      float th = thS[i];
      float sp = (vv > th) ? 1.f : 0.f;
      thS[i] = (th + TH_AMP) * sp + fmaxf(th * TH_DECAY, BASE_TH) * (1.f - sp);
      crS[i] = cc; vrS[i] = vv;
      SR[i] = (_Float16)sp;
    }
    gsync(bar, p.nwg);

    // L: f1 WMMA GEMM 32 x 256 x 256
    gemm(SR, 256, BTF1, 256, YF, B, 256, 256, wid, nw);
    gsync(bar, p.nwg);

    // M: LIF f
    for (int i = gtid; i < ND; i += nth) {
      int o = i & 255;
      float cc = f1cd[o] * cfS[i] + YF[i] + f1b[o];
      cc *= f1m[i];
      float so = (float)SF[i];
      float vv = f1vd[o] * vfS[i] * (1.f - so) + cc;
      cfS[i] = cc; vfS[i] = vv;
      SF[i] = (_Float16)((vv > VTH) ? 1.f : 0.f);
    }
    gsync(bar, p.nwg);

    // N: fc (256->4) + ts_weights accumulation (128 lanes)
    if (gtid < B * NCLASS) {
      int b = gtid >> 2, o = gtid & 3;
      float a = fcb[o];
      for (int k = 0; k < 256; ++k)
        a += (float)SF[b * 256 + k] * fcw[o * 256 + k];
      ACC[gtid] += tsw[t] * a;
    }
    gsync(bar, p.nwg);
  }

  if (gtid < B * NCLASS) p.out[gtid] = ACC[gtid];
}

extern "C" void kernel_launch(void* const* d_in, const int* in_sizes, int n_in,
                              void* d_out, int out_size, void* d_ws,
                              size_t ws_size, hipStream_t stream) {
  (void)in_sizes; (void)out_size; (void)ws_size;
  SnnParams p{};
  int n = n_in < 33 ? n_in : 33;
  for (int i = 0; i < n; ++i) p.in[i] = (const float*)d_in[i];
  p.out = (float*)d_out;
  p.ws = (char*)d_ws;
  const int NWG = 32;  // 32 persistent WGs (one per WGP) x 8 wave32
  p.nwg = NWG;
  // zero the grid-barrier counters (graph-capturable memset node)
  (void)hipMemsetAsync((char*)d_ws + snn::O_SYNC, 0, 64, stream);
  snn_persistent<<<NWG, 256, 0, stream>>>(p);
}
